// GatedGCNLayer_33586644255160
// MI455X (gfx1250) — matrix operand order repaired and checked by reading, not compiled, not run on hardware
//
#include <hip/hip_runtime.h>
#include <hip/hip_bf16.h>

typedef __attribute__((ext_vector_type(16))) _Float16 v16h;
typedef __attribute__((ext_vector_type(8)))  _Float16 v8h;
typedef __attribute__((ext_vector_type(4)))  _Float16 h4;
typedef __attribute__((ext_vector_type(8)))  float    v8f;
typedef __attribute__((ext_vector_type(4)))  float    f4;

#define WMMA_F16(a, b, c) \
  __builtin_amdgcn_wmma_f32_16x16x32_f16(false, (a), false, (b), (short)0, (c), false, false)

#define CONCAT16(lo, hi) \
  __builtin_shufflevector((lo), (hi), 0, 1, 2, 3, 4, 5, 6, 7, 8, 9, 10, 11, 12, 13, 14, 15)

// ---------------------------------------------------------------------------
// WMMA fragment layouts (CDNA5 ISA 7.12.2, wave32)
// A (16x32 f16): lanes 0-15: M=lane, halves0-7 -> K 0..7, halves8-15 -> K16..23
//                lanes16-31: M=lane-16, halves -> K 8..15 / 24..31
//   Row-major LDS tile (stride mult. of 8 halves): two contiguous 16B runs.
// B (32x16 f16): same K order as A: lane = N + 16*((K>>3)&1), half j -> K = (j&7) + 16*(j>>3) + 8*(lane>>4).
//   Staged pre-swizzled so each lane's 16 halves are contiguous (32B).
// D/C f32: VGPR r -> M = r + ((lane&16)?8:0), N = lane&15.
// ---------------------------------------------------------------------------
static __device__ __forceinline__ v16h load_frag_a(const _Float16* __restrict__ tile,
                                                   int row_stride, int kbase) {
  const int lane = threadIdx.x & 31;
  const int r = lane & 15;
  const int k0 = kbase + ((lane & 16) ? 8 : 0);
  const _Float16* p = tile + r * row_stride + k0;
  const v8h lo = *(const v8h*)(p);
  const v8h hi = *(const v8h*)(p + 16);
  return CONCAT16(lo, hi);
}

static __device__ __forceinline__ v16h load_frag_b_sw(const _Float16* __restrict__ wsw,
                                                      int nct, int kt, int ct) {
  const int lane = threadIdx.x & 31;
  const _Float16* p = wsw + (((kt * nct + ct) * 32 + lane) << 4);
  const v8h lo = *(const v8h*)(p);
  const v8h hi = *(const v8h*)(p + 8);
  return CONCAT16(lo, hi);
}

// Scatter weights (row-major [K][cols] f32) into fragment-swizzled f16 LDS.
static __device__ __forceinline__ void stage_weights_sw(const float* __restrict__ W,
                                                        _Float16* __restrict__ wsw,
                                                        int K, int cols, int nthreads) {
  const int nct = cols >> 4;
  for (int i = threadIdx.x; i < K * cols; i += nthreads) {
    const int k = i / cols;  // cols is a compile-time constant at call sites
    const int c = i - k * cols;
    const int kt = k >> 5, kl = k & 31;
    const int ct = c >> 4, cl = c & 15;
    const int lane = cl | ((kl & 8) << 1);
    const int j = (kl & 7) | ((kl & 16) >> 1);
    wsw[(((kt * nct + ct) * 32 + lane) << 4) + j] = (_Float16)W[i];
  }
}

// Stage a [tile_rows x 64] slab of x with BN applied, f32 -> f16, vectorized.
static __device__ __forceinline__ void stage_x_bn(const float* __restrict__ x, int rows,
                                                  int rowbase, int tile_rows,
                                                  const float* __restrict__ scale,
                                                  const float* __restrict__ shift,
                                                  _Float16* __restrict__ xt, int nthreads) {
  for (int i = threadIdx.x; i < tile_rows * 16; i += nthreads) {
    const int r = i >> 4;
    const int c4 = (i & 15) << 2;
    const int gr = rowbase + r;
    f4 v = {0.f, 0.f, 0.f, 0.f};
    if (gr < rows) v = *(const f4*)(x + (size_t)gr * 64 + c4);
    const f4 sc = *(const f4*)(scale + c4);
    const f4 sh = *(const f4*)(shift + c4);
    h4 o;
#pragma unroll
    for (int q = 0; q < 4; ++q) o[q] = (_Float16)(v[q] * sc[q] + sh[q]);
    *(h4*)(xt + r * 72 + c4) = o;
  }
}

// ---------------------------------------------------------------------------
// Utility kernels
// ---------------------------------------------------------------------------
__global__ void zero_f32(float* __restrict__ p, long n) {
  long i = (long)blockIdx.x * blockDim.x + threadIdx.x;
  const long stride = (long)gridDim.x * blockDim.x;
  for (; i < n; i += stride) p[i] = 0.f;
}

__global__ __launch_bounds__(256) void col_stats_partial(const float* __restrict__ x,
                                                         int rows,
                                                         float* __restrict__ partial) {
  __shared__ float s[256];
  __shared__ float q[256];
  const int c = threadIdx.x & 63;
  const int g = threadIdx.x >> 6;
  float sum = 0.f, sq = 0.f;
  for (int r = blockIdx.x * 4 + g; r < rows; r += gridDim.x * 4) {
    const float v = x[(long)r * 64 + c];
    sum += v;
    sq += v * v;
  }
  s[threadIdx.x] = sum;
  q[threadIdx.x] = sq;
  __syncthreads();
  if (g == 0) {
    sum = s[c] + s[64 + c] + s[128 + c] + s[192 + c];
    sq = q[c] + q[64 + c] + q[128 + c] + q[192 + c];
    partial[blockIdx.x * 128 + c] = sum;
    partial[blockIdx.x * 128 + 64 + c] = sq;
  }
}

__global__ void col_stats_final(const float* __restrict__ partial, int nblocks, float inv_n,
                                const float* __restrict__ gamma, const float* __restrict__ beta,
                                float* __restrict__ scale, float* __restrict__ shift) {
  const int c = threadIdx.x;
  if (c >= 64) return;
  float s = 0.f, q = 0.f;
  for (int i = 0; i < nblocks; ++i) {
    s += partial[i * 128 + c];
    q += partial[i * 128 + 64 + c];
  }
  const float mean = s * inv_n;
  const float var = fmaxf(q * inv_n - mean * mean, 0.f);
  const float sc = gamma[c] * rsqrtf(var + 1e-5f);
  scale[c] = sc;
  shift[c] = beta[c] - mean * sc;
}

// ---------------------------------------------------------------------------
// Node GEMMs: [rows x 64] x {Aw,Bw,Cw,Dw} + bias, BN fused into A-staging.
// ---------------------------------------------------------------------------
__global__ __launch_bounds__(256) void node_gemm4(
    const float* __restrict__ h, int rows,
    const float* __restrict__ scale, const float* __restrict__ shift,
    const float* __restrict__ Aw, const float* __restrict__ Bw,
    const float* __restrict__ Cw, const float* __restrict__ Dw,
    const float* __restrict__ Ab, const float* __restrict__ Bb,
    const float* __restrict__ Cb, const float* __restrict__ Db,
    float* __restrict__ Ah, float* __restrict__ Bh,
    float* __restrict__ Ch, float* __restrict__ Dh) {
  __shared__ __align__(16) _Float16 xt[128 * 72];
  __shared__ __align__(16) _Float16 wt[4 * 64 * 64];
  const int t = threadIdx.x;
  const int rowbase = blockIdx.x * 128;

  stage_x_bn(h, rows, rowbase, 128, scale, shift, xt, 256);
  const float* const Ws[4] = {Aw, Bw, Cw, Dw};
#pragma unroll
  for (int m = 0; m < 4; ++m) stage_weights_sw(Ws[m], wt + m * 4096, 64, 64, 256);
  __syncthreads();

  const int wave = t >> 5, lane = t & 31;
  const v16h a0 = load_frag_a(xt + wave * 16 * 72, 72, 0);
  const v16h a1 = load_frag_a(xt + wave * 16 * 72, 72, 32);
  const float* const Bs[4] = {Ab, Bb, Cb, Db};
  float* const Os[4] = {Ah, Bh, Ch, Dh};
  const int col = lane & 15;
  const int rofs = (lane & 16) ? 8 : 0;
  const int rtile = rowbase + wave * 16 + rofs;
  const bool full = (rowbase + 128) <= rows;

#pragma unroll
  for (int m = 0; m < 4; ++m) {
#pragma unroll
    for (int ct = 0; ct < 4; ++ct) {
      const int cg = ct * 16 + col;
      const float bias = Bs[m][cg];
      v8f acc;
#pragma unroll
      for (int j = 0; j < 8; ++j) acc[j] = bias;
      acc = WMMA_F16(a0, load_frag_b_sw(wt + m * 4096, 4, 0, ct), acc);
      acc = WMMA_F16(a1, load_frag_b_sw(wt + m * 4096, 4, 1, ct), acc);
      float* const p = Os[m] + (size_t)rtile * 64 + cg;
      if (full) {
#pragma unroll
        for (int j = 0; j < 8; ++j) p[j * 64] = acc[j];
      } else {
#pragma unroll
        for (int j = 0; j < 8; ++j)
          if (rtile + j < rows) p[j * 64] = acc[j];
      }
    }
  }
}

// ---------------------------------------------------------------------------
// Edge kernel: Ee = bn(e) @ Ew + Eb (WMMA); epilogue: e_new = Ch[src]+Dh[dst]+Ee,
// sigmoid, scatter-adds to sum_sigma_h / sum_sigma, e2 = e + e_new.
// ---------------------------------------------------------------------------
__global__ __launch_bounds__(256) void edge_fused(
    const float* __restrict__ e, int nE,
    const float* __restrict__ scale, const float* __restrict__ shift,
    const float* __restrict__ Ew, const float* __restrict__ Eb,
    const int* __restrict__ src, const int* __restrict__ dst,
    const float* __restrict__ Bh, const float* __restrict__ Ch,
    const float* __restrict__ Dh,
    float* __restrict__ ssh, float* __restrict__ ss, float* __restrict__ e2) {
  __shared__ __align__(16) _Float16 et[128 * 72];
  __shared__ __align__(16) _Float16 wt[64 * 64];
  __shared__ int sI[128];
  __shared__ int dI[128];
  const int t = threadIdx.x;
  const int base = blockIdx.x * 128;

  stage_x_bn(e, nE, base, 128, scale, shift, et, 256);
  stage_weights_sw(Ew, wt, 64, 64, 256);
  if (t < 128) {
    const long ge = base + t;
    sI[t] = (ge < nE) ? src[ge] : 0;
    dI[t] = (ge < nE) ? dst[ge] : 0;
  }
  __syncthreads();

  const int wave = t >> 5, lane = t & 31;
  const v16h a0 = load_frag_a(et + wave * 16 * 72, 72, 0);
  const v16h a1 = load_frag_a(et + wave * 16 * 72, 72, 32);
  const int col = lane & 15;
  const int rofs = (lane & 16) ? 8 : 0;
  const int rl0 = wave * 16 + rofs;
  const long ge0 = base + rl0;
  const bool full = (base + 128) <= nE;

#pragma unroll
  for (int ct = 0; ct < 4; ++ct) {
    const int cg = ct * 16 + col;
    const float bias = Eb[cg];
    v8f acc;
#pragma unroll
    for (int j = 0; j < 8; ++j) acc[j] = bias;
    acc = WMMA_F16(a0, load_frag_b_sw(wt, 4, 0, ct), acc);
    acc = WMMA_F16(a1, load_frag_b_sw(wt, 4, 1, ct), acc);

    const float* const pe = e + ge0 * 64 + cg;
    float* const pe2 = e2 + ge0 * 64 + cg;
    if (full) {
#pragma unroll
      for (int j = 0; j < 8; ++j) {
        const long s = sI[rl0 + j], d = dI[rl0 + j];
        const float enew = acc[j] + Ch[s * 64 + cg] + Dh[d * 64 + cg];
        const float sig = 1.f / (1.f + __expf(-enew));
        atomicAdd(&ssh[d * 64 + cg], Bh[s * 64 + cg] * sig);
        atomicAdd(&ss[d * 64 + cg], sig);
        pe2[j * 64] = pe[j * 64] + enew;
      }
    } else {
#pragma unroll
      for (int j = 0; j < 8; ++j) {
        if (ge0 + j < nE) {
          const long s = sI[rl0 + j], d = dI[rl0 + j];
          const float enew = acc[j] + Ch[s * 64 + cg] + Dh[d * 64 + cg];
          const float sig = 1.f / (1.f + __expf(-enew));
          atomicAdd(&ssh[d * 64 + cg], Bh[s * 64 + cg] * sig);
          atomicAdd(&ss[d * 64 + cg], sig);
          pe2[j * 64] = pe[j * 64] + enew;
        }
      }
    }
  }
}

// h2 = h + Ah + sum_sigma_h / (sum_sigma + eps)
__global__ void node_update(const float* __restrict__ h, const float* __restrict__ Ah,
                            const float* __restrict__ ssh, const float* __restrict__ ss,
                            float* __restrict__ h2, long n) {
  long i = (long)blockIdx.x * blockDim.x + threadIdx.x;
  const long stride = (long)gridDim.x * blockDim.x;
  for (; i < n; i += stride) h2[i] = h[i] + Ah[i] + ssh[i] / (ss[i] + 1e-10f);
}

// ---------------------------------------------------------------------------
// Fused FFN: out = x + relu(bn(x) @ W1 + b1) @ W2 + b2
// Block = 128 threads (4 waves), 64 rows; Z kept in LDS as f16.
// ---------------------------------------------------------------------------
__global__ __launch_bounds__(128) void ffn_fused(
    const float* __restrict__ x, int rows,
    const float* __restrict__ scale, const float* __restrict__ shift,
    const float* __restrict__ W1, const float* __restrict__ b1,
    const float* __restrict__ W2, const float* __restrict__ b2,
    float* __restrict__ out) {
  __shared__ __align__(16) _Float16 xt[64 * 72];
  __shared__ __align__(16) _Float16 w1t[64 * 128];
  __shared__ __align__(16) _Float16 w2t[128 * 64];
  __shared__ __align__(16) _Float16 zt[64 * 136];
  __shared__ float b1s[128];
  const int t = threadIdx.x;
  const int rowbase = blockIdx.x * 64;

  stage_x_bn(x, rows, rowbase, 64, scale, shift, xt, 128);
  stage_weights_sw(W1, w1t, 64, 128, 128);
  stage_weights_sw(W2, w2t, 128, 64, 128);
  if (t < 128) b1s[t] = b1[t];
  __syncthreads();

  const int wave = t >> 5, lane = t & 31;
  const int col = lane & 15;
  const int rofs = (lane & 16) ? 8 : 0;
  const v16h a0 = load_frag_a(xt + wave * 16 * 72, 72, 0);
  const v16h a1 = load_frag_a(xt + wave * 16 * 72, 72, 32);
  _Float16* const zw = zt + wave * 16 * 136;

  // Stage 1: [16x64] @ [64x128] + b1, ReLU -> Z (f16, LDS, row-major stride 136)
#pragma unroll
  for (int ct = 0; ct < 8; ++ct) {
    const float bias = b1s[ct * 16 + col];
    v8f acc;
#pragma unroll
    for (int j = 0; j < 8; ++j) acc[j] = bias;
    acc = WMMA_F16(a0, load_frag_b_sw(w1t, 8, 0, ct), acc);
    acc = WMMA_F16(a1, load_frag_b_sw(w1t, 8, 1, ct), acc);
    _Float16* const zp = zw + rofs * 136 + ct * 16 + col;
#pragma unroll
    for (int j = 0; j < 8; ++j) zp[j * 136] = (_Float16)fmaxf(acc[j], 0.f);
  }
  __syncthreads();

  // Stage 2: [16x128] @ [128x64] + b2 + residual
  const v16h z0 = load_frag_a(zw, 136, 0);
  const v16h z1 = load_frag_a(zw, 136, 32);
  const v16h z2 = load_frag_a(zw, 136, 64);
  const v16h z3 = load_frag_a(zw, 136, 96);
  const int rtile = rowbase + wave * 16 + rofs;
  const bool full = (rowbase + 64) <= rows;
#pragma unroll
  for (int ct = 0; ct < 4; ++ct) {
    const int cg = ct * 16 + col;
    const float bias = b2[cg];
    v8f acc;
#pragma unroll
    for (int j = 0; j < 8; ++j) acc[j] = bias;
    acc = WMMA_F16(z0, load_frag_b_sw(w2t, 4, 0, ct), acc);
    acc = WMMA_F16(z1, load_frag_b_sw(w2t, 4, 1, ct), acc);
    acc = WMMA_F16(z2, load_frag_b_sw(w2t, 4, 2, ct), acc);
    acc = WMMA_F16(z3, load_frag_b_sw(w2t, 4, 3, ct), acc);
    const float* const px = x + (size_t)rtile * 64 + cg;
    float* const po = out + (size_t)rtile * 64 + cg;
    if (full) {
#pragma unroll
      for (int j = 0; j < 8; ++j) po[j * 64] = px[j * 64] + acc[j];
    } else {
#pragma unroll
      for (int j = 0; j < 8; ++j)
        if (rtile + j < rows) po[j * 64] = px[j * 64] + acc[j];
    }
  }
}

// ---------------------------------------------------------------------------
extern "C" void kernel_launch(void* const* d_in, const int* in_sizes, int n_in,
                              void* d_out, int out_size, void* d_ws, size_t ws_size,
                              hipStream_t stream) {
  (void)n_in; (void)out_size; (void)ws_size;
  const float* h   = (const float*)d_in[0];
  const float* e   = (const float*)d_in[1];
  const int*   src = (const int*)d_in[2];
  const int*   dst = (const int*)d_in[3];
  const float* Aw = (const float*)d_in[4];
  const float* Ab = (const float*)d_in[5];
  const float* Bw = (const float*)d_in[6];
  const float* Bb = (const float*)d_in[7];
  const float* Cw = (const float*)d_in[8];
  const float* Cb = (const float*)d_in[9];
  const float* Dw = (const float*)d_in[10];
  const float* Db = (const float*)d_in[11];
  const float* Ew = (const float*)d_in[12];
  const float* Eb = (const float*)d_in[13];
  const float* g1h = (const float*)d_in[14];
  const float* g1e = (const float*)d_in[15];
  const float* g2h = (const float*)d_in[16];
  const float* g2e = (const float*)d_in[17];
  const float* b1h = (const float*)d_in[18];
  const float* b1e = (const float*)d_in[19];
  const float* b2h = (const float*)d_in[20];
  const float* b2e = (const float*)d_in[21];
  const float* Wh1 = (const float*)d_in[22];
  const float* bh1 = (const float*)d_in[23];
  const float* Wh2 = (const float*)d_in[24];
  const float* bh2 = (const float*)d_in[25];
  const float* We1 = (const float*)d_in[26];
  const float* be1 = (const float*)d_in[27];
  const float* We2 = (const float*)d_in[28];
  const float* be2 = (const float*)d_in[29];

  const int N = in_sizes[0] / 64;
  const int E = in_sizes[2];
  const long N64 = (long)N * 64;
  const long E64 = (long)E * 64;

  float* ws = (float*)d_ws;
  float* Ah  = ws; ws += N64;
  float* Bhx = ws; ws += N64;
  float* Chx = ws; ws += N64;
  float* Dhx = ws; ws += N64;
  float* ssh = ws; ws += N64;   // sum_sigma_h
  float* ss  = ws; ws += N64;   // sum_sigma (contiguous with ssh -> one zero pass)
  float* h2  = ws; ws += N64;
  float* e2  = ws; ws += E64;
  const int SB = 512;
  float* partA = ws; ws += (long)SB * 128;
  float* partB = ws; ws += (long)SB * 128;
  float* sc1h = ws; ws += 64;  float* sh1h = ws; ws += 64;
  float* sc1e = ws; ws += 64;  float* sh1e = ws; ws += 64;
  float* sc2h = ws; ws += 64;  float* sh2h = ws; ws += 64;
  float* sc2e = ws; ws += 64;  float* sh2e = ws; ws += 64;

  float* outh = (float*)d_out;
  float* oute = outh + N64;

  // Zero segment-sum accumulators.
  zero_f32<<<1024, 256, 0, stream>>>(ssh, 2 * N64);

  // BN1 stats -> scale/shift
  col_stats_partial<<<SB, 256, 0, stream>>>(h, N, partA);
  col_stats_final<<<1, 64, 0, stream>>>(partA, SB, 1.f / (float)N, g1h, b1h, sc1h, sh1h);
  col_stats_partial<<<SB, 256, 0, stream>>>(e, E, partB);
  col_stats_final<<<1, 64, 0, stream>>>(partB, SB, 1.f / (float)E, g1e, b1e, sc1e, sh1e);

  // Node GEMMs (BN fused)
  node_gemm4<<<(N + 127) / 128, 256, 0, stream>>>(h, N, sc1h, sh1h,
                                                  Aw, Bw, Cw, Dw, Ab, Bb, Cb, Db,
                                                  Ah, Bhx, Chx, Dhx);

  // Edge GEMM + gate + scatter
  edge_fused<<<(E + 127) / 128, 256, 0, stream>>>(e, E, sc1e, sh1e, Ew, Eb, src, dst,
                                                  Bhx, Chx, Dhx, ssh, ss, e2);

  // Node residual update
  node_update<<<1024, 256, 0, stream>>>(h, Ah, ssh, ss, h2, N64);

  // BN2 stats
  col_stats_partial<<<SB, 256, 0, stream>>>(h2, N, partA);
  col_stats_final<<<1, 64, 0, stream>>>(partA, SB, 1.f / (float)N, g2h, b2h, sc2h, sh2h);
  col_stats_partial<<<SB, 256, 0, stream>>>(e2, E, partB);
  col_stats_final<<<1, 64, 0, stream>>>(partB, SB, 1.f / (float)E, g2e, b2e, sc2e, sh2e);

  // Fused FFN + residual -> outputs
  ffn_fused<<<(N + 63) / 64, 128, 0, stream>>>(h2, N, sc2h, sh2h, Wh1, bh1, Wh2, bh2, outh);
  ffn_fused<<<(E + 63) / 64, 128, 0, stream>>>(e2, E, sc2e, sh2e, We1, be1, We2, be2, oute);
}
